// DeformConvBlock_87660282511811
// MI455X (gfx1250) — matrix operand order repaired and checked by reading, not compiled, not run on hardware
//
#include <hip/hip_runtime.h>

// ---------------------------------------------------------------------------
// DeformConvBlock for MI455X (gfx1250, wave32, WMMA)
// x:[1,16,64,64,64] -> two {offset-conv -> deform-conv -> BN+ReLU} layers.
// GEMM contractions run on v_wmma_f32_16x16x32_f16 (f16 in, f32 accumulate).
// ---------------------------------------------------------------------------

typedef __attribute__((ext_vector_type(16))) _Float16 v16h;
typedef __attribute__((ext_vector_type(8)))  float    v8f;

#define NVOX (64 * 64 * 64)   // 2^18 voxels
#define K27  27

// ---- WMMA fragment loaders (wave32; contiguous k = 16*(lane/16) + s in both A and B) ------

// A: 16x32 f16. lanes 0-15 row M=lane, K = 0..15; lanes 16-31 same
// row, K = 16..31.  a_tile layout: [16][Kpad] halfs.
__device__ __forceinline__ v16h load_a_frag(const _Float16* a_tile, int Kpad,
                                            int lane, int k0) {
  int m  = lane & 15;
  int kb = (lane < 16) ? 0 : 16;
  const _Float16* p = a_tile + m * Kpad + k0 + kb;
  v16h a;
#pragma unroll
  for (int j = 0; j < 8; ++j) a[j] = p[j];
#pragma unroll
  for (int j = 0; j < 8; ++j) a[8 + j] = p[8 + j];
  return a;
}

// B: 32x16 f16 (KxN). lanes 0-15 hold K=k0..k0+15 of column N=lane;
// lanes 16-31 hold K=k0+16..k0+31 of column N=lane-16.
// b_tile layout: [numCols][Kpad] halfs (contiguous in K -> 2x b128 ds loads).
__device__ __forceinline__ v16h load_b_frag(const _Float16* b_tile, int Kpad,
                                            int lane, int col, int k0) {
  int kb = (lane < 16) ? 0 : 16;
  const _Float16* p = b_tile + col * Kpad + k0 + kb;
  v16h b;
#pragma unroll
  for (int j = 0; j < 16; ++j) b[j] = p[j];
  return b;
}

__device__ __forceinline__ void wave_lds_fence() {
  __builtin_amdgcn_wave_barrier();
  __asm__ volatile("s_wait_dscnt 0" ::: "memory");
  __builtin_amdgcn_wave_barrier();
}

// ---------------------------------------------------------------------------
// Offset conv: regular 3^3 SAME conv, C in -> 81 out, via im2col + WMMA.
// x:[C][NVOX], w:[81][C*27], bias:[81], out:[81][NVOX]
// ---------------------------------------------------------------------------
template <int C>
__global__ void offset_conv_kernel(const float* __restrict__ x,
                                   const float* __restrict__ w,
                                   const float* __restrict__ bias,
                                   float* __restrict__ out) {
  constexpr int KD  = C * K27;
  constexpr int KP  = (KD + 31) & ~31;
  constexpr int OC  = 81;
  constexpr int OCP = 96;           // 6 N-tiles of 16
  constexpr int NT  = OCP / 16;

  extern __shared__ _Float16 lds[];
  _Float16* b_tile = lds;                               // [OCP][KP]
  const int wave = threadIdx.x >> 5;
  const int lane = threadIdx.x & 31;
  _Float16* a_tile = lds + OCP * KP + wave * 16 * KP;   // wave-private [16][KP]

  // Stage weights f32->f16, zero-pad K rows and out-channel cols.
  for (int i = threadIdx.x; i < OCP * KP; i += blockDim.x) {
    int o = i / KP, k = i - o * KP;
    float v = (o < OC && k < KD) ? w[o * KD + k] : 0.f;
    b_tile[i] = (_Float16)v;
  }
  __syncthreads();

  const int wavesTotal = gridDim.x * (blockDim.x >> 5);
  const int waveId     = blockIdx.x * (blockDim.x >> 5) + wave;
  const int numTiles   = NVOX / 16;

  for (int tile = waveId; tile < numTiles; tile += wavesTotal) {
    const int vbase = tile * 16;

    // im2col: 16 voxels x 27 taps, zero padding at borders.
    for (int p = lane; p < 16 * K27; p += 32) {
      int v = p / K27, t = p - v * K27;
      int vox = vbase + v;
      int wz = vox >> 12, wy = (vox >> 6) & 63, wx = vox & 63;
      int z = wz + t / 9 - 1;
      int y = wy + (t / 3) % 3 - 1;
      int xx = wx + t % 3 - 1;
      bool in = ((unsigned)z < 64u) & ((unsigned)y < 64u) & ((unsigned)xx < 64u);
      int idx = in ? ((z << 6 | y) << 6 | xx) : 0;
      _Float16* arow = a_tile + v * KP + t;
#pragma unroll 4
      for (int c = 0; c < C; ++c) {
        float val = x[c * NVOX + idx];
        arow[c * K27] = in ? (_Float16)val : (_Float16)0.f;
      }
    }
    for (int j = lane; j < 16 * (KP - KD); j += 32) {
      int v = j / (KP - KD), k = KD + (j - v * (KP - KD));
      a_tile[v * KP + k] = (_Float16)0.f;
    }
    wave_lds_fence();

    const int n = lane & 15;
    v8f acc[NT];
#pragma unroll
    for (int nt = 0; nt < NT; ++nt) {
      float bv = (nt * 16 + n < OC) ? bias[nt * 16 + n] : 0.f;
#pragma unroll
      for (int g = 0; g < 8; ++g) acc[nt][g] = bv;
    }

    for (int k0 = 0; k0 < KP; k0 += 32) {
      v16h a = load_a_frag(a_tile, KP, lane, k0);
#pragma unroll
      for (int nt = 0; nt < NT; ++nt) {
        v16h b = load_b_frag(b_tile, KP, lane, nt * 16 + n, k0);
        acc[nt] = __builtin_amdgcn_wmma_f32_16x16x32_f16(
            false, a, false, b, (short)0, acc[nt], false, false);
      }
    }

    const int mbase = (lane < 16) ? 0 : 8;
#pragma unroll
    for (int nt = 0; nt < NT; ++nt) {
      int ch = nt * 16 + n;
      if (ch < OC) {
#pragma unroll
        for (int g = 0; g < 8; ++g)
          out[ch * NVOX + vbase + mbase + g] = acc[nt][g];
      }
    }
    wave_lds_fence();  // A-tile reads retired before next iteration overwrites
  }
}

// ---------------------------------------------------------------------------
// Deformable conv: trilinear-sample 27 taps per voxel per channel into an
// f16 A-tile, contract with w via WMMA.
// x:[C][NVOX], off:[81][NVOX] (dim-major: ch = dim*27+tap),
// w:[32][C*27], bias:[32], out:[32][NVOX]
// ---------------------------------------------------------------------------
template <int C>
__global__ void deform_conv_kernel(const float* __restrict__ x,
                                   const float* __restrict__ off,
                                   const float* __restrict__ w,
                                   const float* __restrict__ bias,
                                   float* __restrict__ out) {
  constexpr int KD = C * K27;
  constexpr int KP = (KD + 31) & ~31;
  constexpr int OC = 32;

  extern __shared__ _Float16 lds[];
  _Float16* b_tile = lds;                              // [32][KP]
  const int wave = threadIdx.x >> 5;
  const int lane = threadIdx.x & 31;
  _Float16* a_tile = lds + OC * KP + wave * 16 * KP;   // wave-private [16][KP]

  for (int i = threadIdx.x; i < OC * KP; i += blockDim.x) {
    int o = i / KP, k = i - o * KP;
    b_tile[i] = (k < KD) ? (_Float16)w[o * KD + k] : (_Float16)0.f;
  }
  __syncthreads();

  const int wavesTotal = gridDim.x * (blockDim.x >> 5);
  const int waveId     = blockIdx.x * (blockDim.x >> 5) + wave;
  const int numTiles   = NVOX / 16;

  for (int tile = waveId; tile < numTiles; tile += wavesTotal) {
    const int vbase = tile * 16;

    // Sample: each lane owns (voxel, tap) pairs; per pair compute the 8
    // trilinear corners once, then gather all C channels.
    for (int p = lane; p < 16 * K27; p += 32) {
      int v = p / K27, t = p - v * K27;
      int vox = vbase + v;
      int wz = vox >> 12, wy = (vox >> 6) & 63, wx = vox & 63;
      float pz = (float)(wz + t / 9 - 1)       + off[(0 * K27 + t) * NVOX + vox];
      float py = (float)(wy + (t / 3) % 3 - 1) + off[(1 * K27 + t) * NVOX + vox];
      float px = (float)(wx + t % 3 - 1)       + off[(2 * K27 + t) * NVOX + vox];
      pz = fminf(fmaxf(pz, 0.f), 63.f);
      py = fminf(fmaxf(py, 0.f), 63.f);
      px = fminf(fmaxf(px, 0.f), 63.f);
      float z0f = floorf(pz), y0f = floorf(py), x0f = floorf(px);
      float fz = pz - z0f, fy = py - y0f, fx = px - x0f;
      int z0 = min(max((int)z0f, 0), 63), z1 = min(z0 + 1, 63);
      int y0 = min(max((int)y0f, 0), 63), y1 = min(y0 + 1, 63);
      int x0 = min(max((int)x0f, 0), 63), x1 = min(x0 + 1, 63);
      int i00 = (z0 * 64 + y0) * 64, i01 = (z0 * 64 + y1) * 64;
      int i10 = (z1 * 64 + y0) * 64, i11 = (z1 * 64 + y1) * 64;
      float gz0 = 1.f - fz, gy0 = 1.f - fy, gx0 = 1.f - fx;
      float w000 = gz0 * gy0 * gx0, w001 = gz0 * gy0 * fx;
      float w010 = gz0 * fy * gx0,  w011 = gz0 * fy * fx;
      float w100 = fz * gy0 * gx0,  w101 = fz * gy0 * fx;
      float w110 = fz * fy * gx0,   w111 = fz * fy * fx;
      _Float16* arow = a_tile + v * KP + t;
#pragma unroll 4
      for (int c = 0; c < C; ++c) {
        const float* xc = x + c * NVOX;
        float s = w000 * xc[i00 + x0] + w001 * xc[i00 + x1]
                + w010 * xc[i01 + x0] + w011 * xc[i01 + x1]
                + w100 * xc[i10 + x0] + w101 * xc[i10 + x1]
                + w110 * xc[i11 + x0] + w111 * xc[i11 + x1];
        arow[c * K27] = (_Float16)s;
      }
    }
    for (int j = lane; j < 16 * (KP - KD); j += 32) {
      int v = j / (KP - KD), k = KD + (j - v * (KP - KD));
      a_tile[v * KP + k] = (_Float16)0.f;
    }
    wave_lds_fence();

    const int n = lane & 15;
    v8f acc0, acc1;
#pragma unroll
    for (int g = 0; g < 8; ++g) { acc0[g] = bias[n]; acc1[g] = bias[16 + n]; }

    for (int k0 = 0; k0 < KP; k0 += 32) {
      v16h a  = load_a_frag(a_tile, KP, lane, k0);
      v16h b0 = load_b_frag(b_tile, KP, lane, n, k0);
      acc0 = __builtin_amdgcn_wmma_f32_16x16x32_f16(
          false, a, false, b0, (short)0, acc0, false, false);
      v16h b1 = load_b_frag(b_tile, KP, lane, 16 + n, k0);
      acc1 = __builtin_amdgcn_wmma_f32_16x16x32_f16(
          false, a, false, b1, (short)0, acc1, false, false);
    }

    const int mbase = (lane < 16) ? 0 : 8;
#pragma unroll
    for (int g = 0; g < 8; ++g) {
      int vox = vbase + mbase + g;
      out[n * NVOX + vox]        = acc0[g];
      out[(16 + n) * NVOX + vox] = acc1[g];
    }
    wave_lds_fence();
  }
}

// ---------------------------------------------------------------------------
// BatchNorm (training-mode batch stats) + ReLU
// ---------------------------------------------------------------------------
__global__ void bn_stats_kernel(const float* __restrict__ raw,
                                float* __restrict__ stats) {
  const int ch = blockIdx.x;
  const float* p = raw + (size_t)ch * NVOX;
  float s1 = 0.f, s2 = 0.f;
  for (int i = threadIdx.x; i < NVOX; i += blockDim.x) {
    float v = p[i];
    s1 += v;
    s2 += v * v;
  }
  __shared__ float sh1[256], sh2[256];
  sh1[threadIdx.x] = s1;
  sh2[threadIdx.x] = s2;
  __syncthreads();
  for (int s = blockDim.x >> 1; s > 0; s >>= 1) {
    if (threadIdx.x < (unsigned)s) {
      sh1[threadIdx.x] += sh1[threadIdx.x + s];
      sh2[threadIdx.x] += sh2[threadIdx.x + s];
    }
    __syncthreads();
  }
  if (threadIdx.x == 0) {
    float mean = sh1[0] * (1.f / (float)NVOX);
    float var  = sh2[0] * (1.f / (float)NVOX) - mean * mean;
    stats[ch]      = mean;
    stats[32 + ch] = rsqrtf(var + 1e-5f);
  }
}

__global__ void bn_apply_kernel(const float* __restrict__ raw,
                                const float* __restrict__ stats,
                                const float* __restrict__ gamma,
                                const float* __restrict__ beta,
                                float* __restrict__ out) {
  int i = blockIdx.x * blockDim.x + threadIdx.x;
  if (i >= 32 * NVOX) return;
  int ch = i >> 18;  // NVOX == 2^18
  float y = (raw[i] - stats[ch]) * stats[32 + ch] * gamma[ch] + beta[ch];
  out[i] = fmaxf(y, 0.f);
}

// ---------------------------------------------------------------------------
extern "C" void kernel_launch(void* const* d_in, const int* in_sizes, int n_in,
                              void* d_out, int out_size, void* d_ws,
                              size_t ws_size, hipStream_t stream) {
  const float* x      = (const float*)d_in[0];
  const float* w_off1 = (const float*)d_in[1];
  const float* b_off1 = (const float*)d_in[2];
  const float* w1     = (const float*)d_in[3];
  const float* b1     = (const float*)d_in[4];
  const float* gamma1 = (const float*)d_in[5];
  const float* beta1  = (const float*)d_in[6];
  const float* w_off2 = (const float*)d_in[7];
  const float* b_off2 = (const float*)d_in[8];
  const float* w2     = (const float*)d_in[9];
  const float* b2     = (const float*)d_in[10];
  const float* gamma2 = (const float*)d_in[11];
  const float* beta2  = (const float*)d_in[12];

  float* ws    = (float*)d_ws;
  float* off   = ws;                          // 81 * NVOX
  float* raw   = off + (size_t)81 * NVOX;     // 32 * NVOX
  float* h     = raw + (size_t)32 * NVOX;     // 32 * NVOX
  float* stats = h + (size_t)32 * NVOX;       // 64

  const dim3 blk(128);   // 4 waves
  const int  grid = 512; // 8 tiles per wave (16384 16-voxel tiles)

  // ---- layer 1 (C = 16, K = 432, Kpad = 448) ----
  {
    constexpr int KP = 448;
    size_t ldsOff = (size_t)(96 * KP + 4 * 16 * KP) * sizeof(_Float16);
    size_t ldsDef = (size_t)(32 * KP + 4 * 16 * KP) * sizeof(_Float16);
    offset_conv_kernel<16><<<grid, blk, ldsOff, stream>>>(x, w_off1, b_off1, off);
    deform_conv_kernel<16><<<grid, blk, ldsDef, stream>>>(x, off, w1, b1, raw);
    bn_stats_kernel<<<32, 256, 0, stream>>>(raw, stats);
    bn_apply_kernel<<<(32 * NVOX + 255) / 256, 256, 0, stream>>>(
        raw, stats, gamma1, beta1, h);
  }
  // ---- layer 2 (C = 32, K = 864, Kpad = 864; LDS sized for 896) ----
  {
    constexpr int KP = 896;
    size_t ldsOff = (size_t)(96 * KP + 4 * 16 * KP) * sizeof(_Float16);
    size_t ldsDef = (size_t)(32 * KP + 4 * 16 * KP) * sizeof(_Float16);
    offset_conv_kernel<32><<<grid, blk, ldsOff, stream>>>(h, w_off2, b_off2, off);
    deform_conv_kernel<32><<<grid, blk, ldsDef, stream>>>(h, off, w2, b2, raw);
    bn_stats_kernel<<<32, 256, 0, stream>>>(raw, stats);
    bn_apply_kernel<<<(32 * NVOX + 255) / 256, 256, 0, stream>>>(
        raw, stats, gamma2, beta2, (float*)d_out);
  }
}
